// Attention_78692390797881
// MI455X (gfx1250) — matrix operand order repaired and checked by reading, not compiled, not run on hardware
//
#include <hip/hip_runtime.h>
#include <hip/hip_bf16.h>
#include <math.h>

typedef __bf16 bf16_t;
typedef __attribute__((ext_vector_type(16))) __bf16 v16bf;
typedef __attribute__((ext_vector_type(8)))  __bf16 v8bf;
typedef __attribute__((ext_vector_type(8)))  float  v8f;
typedef int v4i __attribute__((vector_size(16)));

#define EMBED   768
#define NHEAD   12
#define HDIM    64
#define BATCH   32
#define SEQ     577
#define SPAD    608                 // 19 * 32, padded key/seq length
#define MROWS   (BATCH * SEQ)       // 18464 = 1154 * 16
#define MTILES  (MROWS / 16)        // 1154
#define QT_PER_BH 37                // ceil(577/16)
#define BLKS_PER_BH 5               // ceil(37/8)
#define KSTEPS  (SPAD / 32)         // 19
#define SCALE_F 0.125f              // 64^-0.5
#define NEGBIG  (-1.0e30f)

// ---------------- CDNA5 async global->LDS copy (ASYNCcnt path) ----------------

#if __has_builtin(__builtin_amdgcn_global_load_async_to_lds_b128)
#define HAS_ASYNC_LDS 1
#else
#define HAS_ASYNC_LDS 0
#endif

typedef __attribute__((address_space(1))) v4i* as1_v4i_ptr;
typedef __attribute__((address_space(3))) v4i* as3_v4i_ptr;

__device__ __forceinline__ void async_copy16(const bf16_t* g, bf16_t* l) {
#if HAS_ASYNC_LDS
  __builtin_amdgcn_global_load_async_to_lds_b128((as1_v4i_ptr)g, (as3_v4i_ptr)l,
                                                 0, 0);
#else
  *(v8bf*)l = *(const v8bf*)g;   // sync fallback: global load + ds store
#endif
}

__device__ __forceinline__ void wait_async() {
#if HAS_ASYNC_LDS
  asm volatile("s_wait_asynccnt 0" ::: "memory");
#endif
}

// ---------------- WMMA helpers (CDNA5 16x16x32 bf16, f32 accum) ----------------

__device__ __forceinline__ v8f wmma_bf16(v16bf a, v16bf b, v8f c) {
  return __builtin_amdgcn_wmma_f32_16x16x32_bf16(false, a, false, b, (short)0, c,
                                                 false, false);
}

// A-fragment 16x32 bf16 from row-major [.., ld] storage.
// ISA layout: lane = m + 16*((k>>3)&1); slots 0..7 -> k = hi*8+0..7,
// slots 8..15 -> k = 16 + hi*8 + 0..7.
__device__ __forceinline__ v16bf load_afrag(const bf16_t* base, int ld, int row0,
                                            int k0, int lane) {
  const int m = lane & 15, hi = lane >> 4;
  const bf16_t* p = base + (size_t)(row0 + m) * ld + (k0 + hi * 8);
  v8bf lo = *(const v8bf*)p;
  v8bf hv = *(const v8bf*)(p + 16);
  v16bf r;
#pragma unroll
  for (int i = 0; i < 8; ++i) { r[i] = lo[i]; r[i + 8] = hv[i]; }
  return r;
}

// B-fragment 32x16 bf16 from n-major ("B-transposed") storage BT[n][k], stride ld.
// ISA layout: lane = n + 16*((k>>3)&1); slots 0..7 -> k = hi*8+0..7,
// slots 8..15 -> k = 16 + hi*8 + 0..7 (same order as A).
__device__ __forceinline__ v16bf load_bfrag(const bf16_t* base, int ld, int n0,
                                            int k0, int lane) {
  const bf16_t* p = base + (size_t)(n0 + (lane & 15)) * ld + (k0 + (lane >> 4) * 8);
  v8bf lo = *(const v8bf*)p;
  v8bf hv = *(const v8bf*)(p + 16);
  v16bf r;
#pragma unroll
  for (int i = 0; i < 8; ++i) { r[i] = lo[i]; r[i + 8] = hv[i]; }
  return r;
}

// Row reductions across the 16-lane group holding one matrix row (C layout).
__device__ __forceinline__ float redmax16(float v) {
#pragma unroll
  for (int m = 1; m < 16; m <<= 1) v = fmaxf(v, __shfl_xor(v, m, 32));
  return v;
}
__device__ __forceinline__ float redsum16(float v) {
#pragma unroll
  for (int m = 1; m < 16; m <<= 1) v += __shfl_xor(v, m, 32);
  return v;
}

// Pipelined 16x64 inner product over K=EMBED: 2x-unrolled k-loop with two
// register sets (ping-pong), so prefetch loads write directly into the set the
// next WMMA consumes -- no cross-set register moves.
__device__ __forceinline__ void gemm_16x64(const bf16_t* __restrict__ A, int mrow,
                                           const bf16_t* __restrict__ WT, int ncol,
                                           int lane, v8f acc[4]) {
  v16bf a0 = load_afrag(A, EMBED, mrow, 0, lane);
  v16bf b0[4];
#pragma unroll
  for (int j = 0; j < 4; ++j) b0[j] = load_bfrag(WT, EMBED, ncol + j * 16, 0, lane);

  for (int kt = 0; kt < EMBED - 64; kt += 64) {
    v16bf a1 = load_afrag(A, EMBED, mrow, kt + 32, lane);
    v16bf b1[4];
#pragma unroll
    for (int j = 0; j < 4; ++j)
      b1[j] = load_bfrag(WT, EMBED, ncol + j * 16, kt + 32, lane);
#pragma unroll
    for (int j = 0; j < 4; ++j) acc[j] = wmma_bf16(a0, b0[j], acc[j]);

    a0 = load_afrag(A, EMBED, mrow, kt + 64, lane);
#pragma unroll
    for (int j = 0; j < 4; ++j)
      b0[j] = load_bfrag(WT, EMBED, ncol + j * 16, kt + 64, lane);
#pragma unroll
    for (int j = 0; j < 4; ++j) acc[j] = wmma_bf16(a1, b1[j], acc[j]);
  }
  // tail: kt = EMBED-64 (halves EMBED-64 and EMBED-32)
  {
    v16bf a1 = load_afrag(A, EMBED, mrow, EMBED - 32, lane);
    v16bf b1[4];
#pragma unroll
    for (int j = 0; j < 4; ++j)
      b1[j] = load_bfrag(WT, EMBED, ncol + j * 16, EMBED - 32, lane);
#pragma unroll
    for (int j = 0; j < 4; ++j) acc[j] = wmma_bf16(a0, b0[j], acc[j]);
#pragma unroll
    for (int j = 0; j < 4; ++j) acc[j] = wmma_bf16(a1, b1[j], acc[j]);
  }
}

// ---------------- conversion kernels ----------------

__global__ __launch_bounds__(256) void cvt_kernel(const float* __restrict__ in,
                                                  bf16_t* __restrict__ out,
                                                  size_t n) {
  size_t i = (size_t)blockIdx.x * 256 + threadIdx.x;
  if (i < n) out[i] = (bf16_t)in[i];
}

// in: [R][C] f32 row-major  ->  out: [C][R] bf16 (n-major for B-fragments)
__global__ __launch_bounds__(256) void transpose_cvt_kernel(
    const float* __restrict__ in, bf16_t* __restrict__ out, int R, int C) {
  size_t i = (size_t)blockIdx.x * 256 + threadIdx.x;
  if (i < (size_t)R * C) {
    int r = (int)(i / C), c = (int)(i % C);
    out[(size_t)c * R + r] = (bf16_t)in[i];
  }
}

// ---------------- QKV GEMM: [18464,768] x [768,2304] + bias ----------------

__global__ __launch_bounds__(256) void qkv_gemm_kernel(
    const bf16_t* __restrict__ Xb, const bf16_t* __restrict__ WT,
    const float* __restrict__ bias, bf16_t* __restrict__ Qb,
    bf16_t* __restrict__ Kb, bf16_t* __restrict__ VTb) {
  const int lane = threadIdx.x & 31;
  const int wave = blockIdx.x * 8 + (threadIdx.x >> 5);
  const int NT64 = (3 * EMBED) / 64;  // 36
  const int mt = wave / NT64;
  const int nt = wave % NT64;
  if (mt >= MTILES) return;
  const int mrow = mt * 16, ncol = nt * 64;

  v8f acc[4] = {};
  gemm_16x64(Xb, mrow, WT, ncol, lane, acc);

  const int hi = lane >> 4, nl = lane & 15;
#pragma unroll
  for (int j = 0; j < 4; ++j) {
    const int n = ncol + j * 16 + nl;           // column in [0, 2304)
    const int which = n / EMBED;                // 0=q 1=k 2=v
    const int rem = n % EMBED;
    const int h = rem >> 6, dh = rem & 63;
    const float bv = bias[n];
#pragma unroll
    for (int r = 0; r < 8; ++r) {
      const int gm = mrow + hi * 8 + r;         // global row
      const int bb = gm / SEQ, s = gm % SEQ;
      const float v = acc[j][r] + bv;
      const size_t bh = (size_t)bb * NHEAD + h;
      if (which == 0)       Qb[(bh * SPAD + s) * HDIM + dh] = (bf16_t)v;
      else if (which == 1)  Kb[(bh * SPAD + s) * HDIM + dh] = (bf16_t)v;
      else                  VTb[(bh * HDIM + dh) * SPAD + s] = (bf16_t)v;
    }
  }
}

// ---------------- Fused flash attention ----------------
// One block (8 waves) owns one (b,h); waves cover 8 q-tiles. K/V tiles are
// staged into LDS once per block via async global->LDS (double-buffered,
// ASYNCcnt + barrier), shared by all 8 waves: 8x less global K/V traffic.

__global__ __launch_bounds__(256) void flash_attn_kernel(
    const bf16_t* __restrict__ Q, const bf16_t* __restrict__ K,
    const bf16_t* __restrict__ VT, bf16_t* __restrict__ AO) {
  __shared__ __align__(32) bf16_t Kt[2][32 * 64];   // 32 keys x 64 dh
  __shared__ __align__(32) bf16_t Vt[2][64 * 32];   // 64 dh x 32 keys (V^T tile)
  __shared__ __align__(32) bf16_t pbuf[8][16 * 40]; // per-wave P C->A relayout
  const int tid = threadIdx.x;
  const int lane = tid & 31;
  const int widx = tid >> 5;
  const int bh = blockIdx.x / BLKS_PER_BH;
  const int blk = blockIdx.x % BLKS_PER_BH;
  int qt = blk * 8 + widx;
  if (qt > QT_PER_BH - 1) qt = QT_PER_BH - 1;   // duplicate tile 36: benign

  const bf16_t* Qp = Q + (size_t)bh * SPAD * HDIM;
  const bf16_t* Kp = K + (size_t)bh * SPAD * HDIM;
  const bf16_t* Vp = VT + (size_t)bh * HDIM * SPAD;
  bf16_t* pb = &pbuf[widx][0];
  const int hi = lane >> 4, nl = lane & 15;

  // cooperative 256-thread tile fill: 16B per thread per tile
  const int krow = tid >> 3, koff = (tid & 7) * 8;  // K: 32 rows x 64
  const int vrow = tid >> 2, voff = (tid & 3) * 8;  // V: 64 rows x 32
  auto fill = [&](int buf, int kb) {
    async_copy16(Kp + (size_t)(kb + krow) * HDIM + koff,
                 &Kt[buf][krow * HDIM + koff]);
    async_copy16(Vp + (size_t)vrow * SPAD + kb + voff,
                 &Vt[buf][vrow * 32 + voff]);
  };

  // Q tile stays in registers across all key steps.
  v16bf qa0 = load_afrag(Qp, HDIM, qt * 16, 0, lane);
  v16bf qa1 = load_afrag(Qp, HDIM, qt * 16, 32, lane);

  v8f o0 = {}, o1 = {}, o2 = {}, o3 = {};       // 16x64 f32 accumulator
  float mrun[8], lrun[8];
#pragma unroll
  for (int r = 0; r < 8; ++r) { mrun[r] = NEGBIG; lrun[r] = 0.0f; }

  fill(0, 0);
  for (int i = 0; i < KSTEPS; ++i) {
    const int kb = i * 32;
    wait_async();        // own async fills done
    __syncthreads();     // everyone's fills for buf i&1 done; prev reads done
    if (i + 1 < KSTEPS) fill((i + 1) & 1, kb + 32);

    const bf16_t* Kl = &Kt[i & 1][0];
    const bf16_t* Vl = &Vt[i & 1][0];

    // scores for 32 keys: two 16x16 C fragments, inner dim Dh=64
    v8f s0 = {}, s1 = {};
    s0 = wmma_bf16(qa0, load_bfrag(Kl, HDIM, 0,  0,  lane), s0);
    s0 = wmma_bf16(qa1, load_bfrag(Kl, HDIM, 0,  32, lane), s0);
    s1 = wmma_bf16(qa0, load_bfrag(Kl, HDIM, 16, 0,  lane), s1);
    s1 = wmma_bf16(qa1, load_bfrag(Kl, HDIM, 16, 32, lane), s1);

    const bool v0 = (kb + nl) < SEQ;
    const bool v1 = (kb + 16 + nl) < SEQ;

    float esc[8];
#pragma unroll
    for (int r = 0; r < 8; ++r) {
      float a0 = v0 ? s0[r] * SCALE_F : NEGBIG;
      float a1 = v1 ? s1[r] * SCALE_F : NEGBIG;
      float rm = redmax16(fmaxf(a0, a1));
      float mn = fmaxf(mrun[r], rm);
      float p0 = __expf(a0 - mn);
      float p1 = __expf(a1 - mn);
      float rs = redsum16(p0 + p1);
      float sc = __expf(mrun[r] - mn);
      lrun[r] = lrun[r] * sc + rs;
      mrun[r] = mn;
      esc[r] = sc;
      const int row = hi * 8 + r;               // C-layout row for this lane group
      pb[row * 40 + nl]      = (bf16_t)p0;      // local keys 0..15
      pb[row * 40 + 16 + nl] = (bf16_t)p1;      // local keys 16..31
    }
#pragma unroll
    for (int r = 0; r < 8; ++r) {
      o0[r] *= esc[r]; o1[r] *= esc[r]; o2[r] *= esc[r]; o3[r] *= esc[r];
    }

    // make all lanes' P stores visible before A-fragment reload (same wave)
    asm volatile("s_wait_dscnt 0" ::: "memory");
    v16bf pa = load_afrag(pb, 40, 0, 0, lane);

    // O += P(16x32) x V(32x64)
    o0 = wmma_bf16(pa, load_bfrag(Vl, 32, 0,  0, lane), o0);
    o1 = wmma_bf16(pa, load_bfrag(Vl, 32, 16, 0, lane), o1);
    o2 = wmma_bf16(pa, load_bfrag(Vl, 32, 32, 0, lane), o2);
    o3 = wmma_bf16(pa, load_bfrag(Vl, 32, 48, 0, lane), o3);
  }

  const int b = bh / NHEAD, h = bh % NHEAD;
#pragma unroll
  for (int r = 0; r < 8; ++r) {
    const int s = qt * 16 + hi * 8 + r;
    if (s < SEQ) {
      const float inv = 1.0f / lrun[r];
      const size_t base = ((size_t)b * SEQ + s) * EMBED + h * HDIM + nl;
      AO[base +  0] = (bf16_t)(o0[r] * inv);
      AO[base + 16] = (bf16_t)(o1[r] * inv);
      AO[base + 32] = (bf16_t)(o2[r] * inv);
      AO[base + 48] = (bf16_t)(o3[r] * inv);
    }
  }
}

// ---------------- FC GEMM: [18464,768] x [768,768] + bias -> f32 out ----------------

__global__ __launch_bounds__(256) void fc_gemm_kernel(
    const bf16_t* __restrict__ Ab, const bf16_t* __restrict__ WT,
    const float* __restrict__ bias, float* __restrict__ out) {
  const int lane = threadIdx.x & 31;
  const int wave = blockIdx.x * 8 + (threadIdx.x >> 5);
  const int NT64 = EMBED / 64;  // 12
  const int mt = wave / NT64;
  const int nt = wave % NT64;
  if (mt >= MTILES) return;
  const int mrow = mt * 16, ncol = nt * 64;

  v8f acc[4] = {};
  gemm_16x64(Ab, mrow, WT, ncol, lane, acc);

  const int hi = lane >> 4, nl = lane & 15;
#pragma unroll
  for (int j = 0; j < 4; ++j) {
    const int n = ncol + j * 16 + nl;
    const float bv = bias[n];
#pragma unroll
    for (int r = 0; r < 8; ++r) {
      const int gm = mrow + hi * 8 + r;
      out[(size_t)gm * EMBED + n] = acc[j][r] + bv;
    }
  }
}

// ---------------- launch ----------------

extern "C" void kernel_launch(void* const* d_in, const int* in_sizes, int n_in,
                              void* d_out, int out_size, void* d_ws, size_t ws_size,
                              hipStream_t stream) {
  const float* x     = (const float*)d_in[0];
  const float* w_qkv = (const float*)d_in[1];
  const float* b_qkv = (const float*)d_in[2];
  const float* w_fc  = (const float*)d_in[3];
  const float* b_fc  = (const float*)d_in[4];
  float* out = (float*)d_out;

  auto al = [](size_t v) { return (v + 255) & ~(size_t)255; };
  char* p = (char*)d_ws;
  bf16_t* xb    = (bf16_t*)p; p += al((size_t)MROWS * EMBED * 2);
  bf16_t* wqkvT = (bf16_t*)p; p += al((size_t)3 * EMBED * EMBED * 2);
  bf16_t* wfcT  = (bf16_t*)p; p += al((size_t)EMBED * EMBED * 2);
  const size_t qkv_elems = (size_t)BATCH * NHEAD * SPAD * HDIM;
  bf16_t* Qb  = (bf16_t*)p; p += al(qkv_elems * 2);
  bf16_t* Kb  = (bf16_t*)p; p += al(qkv_elems * 2);
  bf16_t* VTb = (bf16_t*)p; p += al(qkv_elems * 2);
  bf16_t* AO  = (bf16_t*)p; p += al((size_t)MROWS * EMBED * 2);

  // zero the padded Q/K/V buffers so pad rows contribute nothing
  (void)hipMemsetAsync(Qb,  0, qkv_elems * 2, stream);
  (void)hipMemsetAsync(Kb,  0, qkv_elems * 2, stream);
  (void)hipMemsetAsync(VTb, 0, qkv_elems * 2, stream);

  const size_t nx = (size_t)MROWS * EMBED;
  cvt_kernel<<<(unsigned)((nx + 255) / 256), 256, 0, stream>>>(x, xb, nx);
  const size_t nq = (size_t)EMBED * 3 * EMBED;
  transpose_cvt_kernel<<<(unsigned)((nq + 255) / 256), 256, 0, stream>>>(
      w_qkv, wqkvT, EMBED, 3 * EMBED);
  const size_t nf = (size_t)EMBED * EMBED;
  transpose_cvt_kernel<<<(unsigned)((nf + 255) / 256), 256, 0, stream>>>(
      w_fc, wfcT, EMBED, EMBED);

  const int qkv_waves = MTILES * ((3 * EMBED) / 64);          // 41544
  qkv_gemm_kernel<<<qkv_waves / 8, 256, 0, stream>>>(xb, wqkvT, b_qkv, Qb, Kb, VTb);

  flash_attn_kernel<<<BATCH * NHEAD * BLKS_PER_BH, 256, 0, stream>>>(Qb, Kb, VTb, AO);

  const int fc_waves = MTILES * (EMBED / 64);                 // 13848
  fc_gemm_kernel<<<fc_waves / 8, 256, 0, stream>>>(AO, wfcT, b_fc, out);
}
